// Unit_KED_43834436223133
// MI455X (gfx1250) — matrix operand order repaired and checked by reading, not compiled, not run on hardware
//
#include <hip/hip_runtime.h>
#include <hip/hip_bf16.h>
#include <math.h>

typedef __attribute__((ext_vector_type(16))) _Float16 v16h;
typedef __attribute__((ext_vector_type(8)))  float    v8f;

#define BATCH 4
#define NPTS  3136     // 56*56
#define DM    256      // GNN d_model
#define BNS   0.99999500003749972f   // 1/sqrt(1+1e-5)

union Frag16 { v16h v; uint4 q[2]; };          // 32B fragment <- two b128 loads
union H8    { uint4 q; _Float16 h[8]; };        // b128 <-> 8 halfs

__device__ __forceinline__ float gelu_exact(float x){
  return 0.5f * x * (1.0f + erff(x * 0.70710678118654752440f));
}

// ---------------- f32 -> f16 convert ----------------
__global__ void f2h_kernel(const float* __restrict__ s, _Float16* __restrict__ d, int n){
  int i = blockIdx.x * blockDim.x + threadIdx.x;
  if (i < n) d[i] = (_Float16)s[i];
}

// ---------- conv3x3 s2 p1 + bias + BN + ReLU : (B,64,224,224)->(B,128,112,112) ----------
__global__ void conv3x3s2_kernel(const float* __restrict__ D, const float* __restrict__ Wc,
                                 const float* __restrict__ bias, const float* __restrict__ g,
                                 const float* __restrict__ be, float* __restrict__ T){
  int t = blockIdx.x * blockDim.x + threadIdx.x;
  const int total = BATCH * 128 * 112 * 112;
  if (t >= total) return;
  int ow = t % 112; int oh = (t / 112) % 112;
  int o  = (t / (112 * 112)) % 128; int b = t / (112 * 112 * 128);
  int ih0 = oh * 2 - 1, iw0 = ow * 2 - 1;
  float acc = 0.f;
  const float* db = D + (size_t)b * 64 * 224 * 224;
  const float* wb = Wc + (size_t)o * 64 * 9;
  for (int i = 0; i < 64; ++i){
    const float* dp = db + (size_t)i * 224 * 224;
    const float* wp = wb + i * 9;
    #pragma unroll
    for (int kh = 0; kh < 3; ++kh){
      int ih = ih0 + kh; if (ih < 0 || ih >= 224) continue;
      #pragma unroll
      for (int kw = 0; kw < 3; ++kw){
        int iw = iw0 + kw; if (iw < 0 || iw >= 224) continue;
        acc = fmaf(dp[ih * 224 + iw], wp[kh * 3 + kw], acc);
      }
    }
  }
  acc += bias[o];
  acc = acc * (g[o] * BNS) + be[o];
  T[t] = fmaxf(acc, 0.f);
}

// ---------- concat + maxpool2x2 : -> x (B,256,3136) f32 + f16 ----------
__global__ void pool_concat_kernel(const float* __restrict__ F, const float* __restrict__ T,
                                   float* __restrict__ Xf, _Float16* __restrict__ Xh){
  int t = blockIdx.x * blockDim.x + threadIdx.x;
  const int total = BATCH * 256 * 56 * 56;
  if (t >= total) return;
  int pw = t % 56; int ph = (t / 56) % 56;
  int c  = (t / (56 * 56)) % 256; int b = t / (56 * 56 * 256);
  const float* src = (c < 128) ? (F + ((size_t)b * 128 + c) * 112 * 112)
                               : (T + ((size_t)b * 128 + (c - 128)) * 112 * 112);
  int r = ph * 2, q = pw * 2;
  float m = fmaxf(fmaxf(src[r * 112 + q], src[r * 112 + q + 1]),
                  fmaxf(src[(r + 1) * 112 + q], src[(r + 1) * 112 + q + 1]));
  size_t o = ((size_t)b * 256 + c) * NPTS + ph * 56 + pw;
  Xf[o] = m; Xh[o] = (_Float16)m;
}

// ---------- generic WMMA GEMM:  Y = epi( W(MxK) * X(KxN) ) ----------
// grid: (N/64, M/128, B); block 256 (8 waves). A-frags straight from global
// (L2-resident weights, vector b128), X tile transposed through LDS so B-frags
// are two ds_load_b128 each. 4 accumulators/wave (128x64 tile).
__global__ void __launch_bounds__(256)
gemm_bn_act(const _Float16* __restrict__ W, const _Float16* __restrict__ X,
            const float* __restrict__ bias, const float* __restrict__ gamma,
            const float* __restrict__ beta, const float* __restrict__ Res,
            float* __restrict__ Yf, _Float16* __restrict__ Yh,
            int K, long long ldXb, long long ldYb, int act){
  __shared__ _Float16 Xt[64][40];           // [n][k] transposed, padded rows (80B, 16B-mult)
  const int tid = threadIdx.x;
  const int w  = tid >> 5;                  // wave 0..7
  const int l  = tid & 31;
  const int lm = l & 15;
  const int hi = (l >> 4) & 1;
  const int tn = blockIdx.x * 64;
  const int tm = blockIdx.y * 128;
  const int b  = blockIdx.z;
  const _Float16* Xb = X + (size_t)b * ldXb;
  const _Float16* Wr = W + (size_t)(tm + (w << 4) + lm) * K;   // this lane's A row

  const int kx = tid >> 3;                  // staging: k row 0..31
  const int n8 = (tid & 7) << 3;            // staging: 8-col group

  v8f acc0 = {}, acc1 = {}, acc2 = {}, acc3 = {};
  for (int kk = 0; kk < K; kk += 32){
    { // stage 32k x 64n tile, transposing via registers (coalesced b128 global read)
      H8 u; u.q = *(const uint4*)(Xb + (size_t)(kk + kx) * NPTS + tn + n8);
      #pragma unroll
      for (int i = 0; i < 8; ++i) Xt[n8 + i][kx] = u.h[i];
    }
    __syncthreads();
    Frag16 a;                                // A layout: one contiguous 16-half run
    a.q[0] = *(const uint4*)(Wr + kk + hi * 16);
    a.q[1] = *(const uint4*)(Wr + kk + hi * 16 + 8);
    if (kk + 32 < K) __builtin_prefetch(Wr + kk + 32, 0, 3);
    Frag16 b0, b1, b2, b3;                   // B layout: one contiguous 16-half run
    b0.q[0] = *(const uint4*)&Xt[lm     ][hi * 16];
    b0.q[1] = *(const uint4*)&Xt[lm     ][hi * 16 + 8];
    b1.q[0] = *(const uint4*)&Xt[lm + 16][hi * 16];
    b1.q[1] = *(const uint4*)&Xt[lm + 16][hi * 16 + 8];
    b2.q[0] = *(const uint4*)&Xt[lm + 32][hi * 16];
    b2.q[1] = *(const uint4*)&Xt[lm + 32][hi * 16 + 8];
    b3.q[0] = *(const uint4*)&Xt[lm + 48][hi * 16];
    b3.q[1] = *(const uint4*)&Xt[lm + 48][hi * 16 + 8];
    acc0 = __builtin_amdgcn_wmma_f32_16x16x32_f16(false, a.v, false, b0.v, (short)0, acc0, false, false);
    acc1 = __builtin_amdgcn_wmma_f32_16x16x32_f16(false, a.v, false, b1.v, (short)0, acc1, false, false);
    acc2 = __builtin_amdgcn_wmma_f32_16x16x32_f16(false, a.v, false, b2.v, (short)0, acc2, false, false);
    acc3 = __builtin_amdgcn_wmma_f32_16x16x32_f16(false, a.v, false, b3.v, (short)0, acc3, false, false);
    __syncthreads();
  }
  // fused epilogue: bias -> BN -> (gelu) -> (+res) -> store f32/f16
  const int n0 = tn + lm;
  #pragma unroll
  for (int v = 0; v < 8; ++v){
    int m = tm + (w << 4) + hi * 8 + v;
    float bi = bias[m];
    float s  = gamma[m] * BNS;
    float bt = beta[m];
    float y0 = (acc0[v] + bi) * s + bt;
    float y1 = (acc1[v] + bi) * s + bt;
    float y2 = (acc2[v] + bi) * s + bt;
    float y3 = (acc3[v] + bi) * s + bt;
    if (act == 1){
      y0 = gelu_exact(y0); y1 = gelu_exact(y1);
      y2 = gelu_exact(y2); y3 = gelu_exact(y3);
    }
    size_t o0 = (size_t)b * ldYb + (size_t)m * NPTS + n0;
    if (Res){ y0 += Res[o0]; y1 += Res[o0+16]; y2 += Res[o0+32]; y3 += Res[o0+48]; }
    if (Yf){ Yf[o0] = y0; Yf[o0+16] = y1; Yf[o0+32] = y2; Yf[o0+48] = y3; }
    if (Yh){
      Yh[o0]    = (_Float16)y0; Yh[o0+16] = (_Float16)y1;
      Yh[o0+32] = (_Float16)y2; Yh[o0+48] = (_Float16)y3;
    }
  }
}

// ---------- L2-normalize: y1 (B,C,N) f32 -> fnT (B,N,C) f16 + sq ----------
// Transposed output makes both Gram-matrix fragment reads contiguous b128.
__global__ void normalize_kernel(const float* __restrict__ Y, _Float16* __restrict__ FT,
                                 float* __restrict__ SQ){
  int t = blockIdx.x * blockDim.x + threadIdx.x;
  if (t >= BATCH * NPTS) return;
  int b = t / NPTS, n = t % NPTS;
  const float* y = Y + (size_t)b * DM * NPTS + n;
  float s2 = 0.f;
  for (int c = 0; c < DM; ++c){ float v = y[(size_t)c * NPTS]; s2 = fmaf(v, v, s2); }
  float inv = 1.0f / fmaxf(sqrtf(s2), 1e-12f);
  _Float16* f = FT + ((size_t)b * NPTS + n) * DM;
  for (int c0 = 0; c0 < DM; c0 += 8){
    H8 u;
    #pragma unroll
    for (int i = 0; i < 8; ++i) u.h[i] = (_Float16)(y[(size_t)(c0 + i) * NPTS] * inv);
    *(uint4*)(f + c0) = u.q;
  }
  SQ[t] = s2 * inv * inv;
}

// ---------- fused WMMA Gram-matrix + streaming top-k (1 wave / 16 rows) ----------
// fnT is (N,C): A and B fragments are direct contiguous b128 global loads (L2-hot).
__global__ void __launch_bounds__(32)
knn_kernel(const _Float16* __restrict__ FT, const float* __restrict__ SQ,
           int* __restrict__ IDX, int k){
  const int b  = blockIdx.y;
  const int na = blockIdx.x * 16;
  const int l  = threadIdx.x;
  const int lm = l & 15;
  const int hi = (l >> 4) & 1;
  const _Float16* fb = FT + (size_t)b * NPTS * DM;

  // A fragments (16 query rows, full K=256) held in registers: 16 b128 loads
  Frag16 A[8];
  const _Float16* ar = fb + (size_t)(na + lm) * DM;
  #pragma unroll
  for (int ch = 0; ch < 8; ++ch){
    A[ch].q[0] = *(const uint4*)(ar + ch * 32 + hi * 16);
    A[ch].q[1] = *(const uint4*)(ar + ch * 32 + hi * 16 + 8);
  }
  __shared__ float sqA[16];
  __shared__ float sqB[16];
  __shared__ float Dt[16][17];
  if (l < 16) sqA[l] = SQ[(size_t)b * NPTS + na + l];

  float tv[8]; int ti[8];
  #pragma unroll
  for (int i = 0; i < 8; ++i){ tv[i] = -3.4e38f; ti[i] = 0; }

  for (int nb = 0; nb < NPTS / 16; ++nb){
    const int n0 = nb * 16;
    __syncthreads();                               // Dt/sqB reuse guard
    if (l < 16) sqB[l] = SQ[(size_t)b * NPTS + n0 + l];
    const _Float16* br = fb + (size_t)(n0 + lm) * DM;
    v8f acc = {};
    #pragma unroll
    for (int ch = 0; ch < 8; ++ch){
      Frag16 bf;
      bf.q[0] = *(const uint4*)(br + ch * 32 + hi * 16);
      bf.q[1] = *(const uint4*)(br + ch * 32 + hi * 16 + 8);
      acc = __builtin_amdgcn_wmma_f32_16x16x32_f16(false, A[ch].v, false, bf.v, (short)0, acc, false, false);
    }
    __syncthreads();
    #pragma unroll
    for (int v = 0; v < 8; ++v){
      int m = hi * 8 + v;
      Dt[m][lm] = 2.0f * acc[v] - sqA[m] - sqB[lm];
    }
    __syncthreads();
    if (l < 16){
      for (int j = 0; j < 16; ++j){
        float val = Dt[l][j]; int id = n0 + j;
        #pragma unroll
        for (int p2 = 0; p2 < 8; ++p2){
          if (val > tv[p2]){
            float tf = tv[p2]; tv[p2] = val; val = tf;
            int   tg = ti[p2]; ti[p2] = id;  id  = tg;
          }
        }
      }
    }
  }
  if (l < 16){
    size_t o = ((size_t)b * NPTS + na + l) * (size_t)k;
    #pragma unroll
    for (int j = 0; j < 8; ++j) if (j < k) IDX[o + j] = ti[j];
  }
}

// ---------- max-relative gather: y1 + idx -> interleaved (B,512,N) f16 ----------
__global__ void mrconv_kernel(const float* __restrict__ X, const int* __restrict__ IDX,
                              _Float16* __restrict__ Y2, int k){
  int t = blockIdx.x * blockDim.x + threadIdx.x;
  const int total = BATCH * DM * NPTS;
  if (t >= total) return;
  int n = t % NPTS; int c = (t / NPTS) % DM; int b = t / (NPTS * DM);
  const float* xb = X + ((size_t)b * DM + c) * NPTS;
  float xc = xb[n];
  const int* id = IDX + ((size_t)b * NPTS + n) * (size_t)k;
  float mx = -3.4e38f;
  for (int j = 0; j < k; ++j) mx = fmaxf(mx, xb[id[j]]);
  _Float16* y = Y2 + ((size_t)b * 512 + 2 * c) * NPTS + n;
  y[0]    = (_Float16)xc;
  y[NPTS] = (_Float16)(mx - xc);
}

extern "C" void kernel_launch(void* const* d_in, const int* in_sizes, int n_in,
                              void* d_out, int out_size, void* d_ws, size_t ws_size,
                              hipStream_t stream){
  const float* feature = (const float*)d_in[0];
  const float* deep    = (const float*)d_in[1];
  const float* cw  = (const float*)d_in[2];
  const float* cb  = (const float*)d_in[3];
  const float* cg  = (const float*)d_in[4];
  const float* cbe = (const float*)d_in[5];

  char* ws = (char*)d_ws;
  size_t off = 0;
  auto alloc = [&](size_t bytes){ size_t o = off; off = (off + bytes + 255) & ~((size_t)255); return o; };

  float*    convT = (float*)   (ws + alloc((size_t)BATCH * 128 * 112 * 112 * 4));
  float*    x_f   = (float*)   (ws + alloc((size_t)BATCH * 256  * NPTS * 4));
  _Float16* x_h   = (_Float16*)(ws + alloc((size_t)BATCH * 256  * NPTS * 2));
  float*    y1_f  = (float*)   (ws + alloc((size_t)BATCH * 256  * NPTS * 4));
  _Float16* fnT_h = (_Float16*)(ws + alloc((size_t)BATCH * NPTS * 256 * 2));
  float*    sq_f  = (float*)   (ws + alloc((size_t)BATCH * NPTS * 4));
  int*      idx_i = (int*)     (ws + alloc((size_t)BATCH * NPTS * 8 * 4));
  _Float16* y2_h  = (_Float16*)(ws + alloc((size_t)BATCH * 512  * NPTS * 2));
  _Float16* gg_h  = (_Float16*)(ws + alloc((size_t)BATCH * 512  * NPTS * 2));
  _Float16* h_h   = (_Float16*)(ws + alloc((size_t)BATCH * 1024 * NPTS * 2));

  const int wsz[5]  = {256 * 256, 512 * 128, 256 * 512, 1024 * 256, 256 * 1024};
  const int widx[5] = {0, 4, 8, 12, 16};
  _Float16* wh[2][5];
  for (int blk = 0; blk < 2; ++blk)
    for (int j = 0; j < 5; ++j)
      wh[blk][j] = (_Float16*)(ws + alloc((size_t)wsz[j] * 2));

  // convert weights to f16 (deterministic, every launch)
  for (int blk = 0; blk < 2; ++blk){
    int base = 6 + blk * 20;
    for (int j = 0; j < 5; ++j)
      f2h_kernel<<<(wsz[j] + 255) / 256, 256, 0, stream>>>(
        (const float*)d_in[base + widx[j]], wh[blk][j], wsz[j]);
  }

  // stem: conv + BN + ReLU, then concat + maxpool (f32 + f16 copies)
  conv3x3s2_kernel<<<(BATCH * 128 * 112 * 112 + 255) / 256, 256, 0, stream>>>(
      deep, cw, cb, cg, cbe, convT);
  pool_concat_kernel<<<(BATCH * 256 * 56 * 56 + 255) / 256, 256, 0, stream>>>(
      feature, convT, x_f, x_h);

  for (int blk = 0; blk < 2; ++blk){
    int base = 6 + blk * 20;
    auto P = [&](int i){ return (const float*)d_in[base + i]; };
    int k = (blk == 0) ? 4 : 8;

    // fc1: 256x256, BN -> y1 f32
    gemm_bn_act<<<dim3(49, 2, BATCH), 256, 0, stream>>>(
        wh[blk][0], x_h, P(1), P(2), P(3), nullptr, y1_f, nullptr,
        256, 256LL * NPTS, 256LL * NPTS, 0);
    // kNN on normalized y1 (transposed f16 features)
    normalize_kernel<<<(BATCH * NPTS + 255) / 256, 256, 0, stream>>>(y1_f, fnT_h, sq_f);
    knn_kernel<<<dim3(NPTS / 16, BATCH), 32, 0, stream>>>(fnT_h, sq_f, idx_i, k);
    // max-relative graph feature, channel-interleaved -> (B,512,N) f16
    mrconv_kernel<<<(BATCH * DM * NPTS + 255) / 256, 256, 0, stream>>>(y1_f, idx_i, y2_h, k);
    // grouped conv (4 independent 128x128 GEMMs), BN + GELU -> gg f16
    for (int g2 = 0; g2 < 4; ++g2){
      gemm_bn_act<<<dim3(49, 1, BATCH), 256, 0, stream>>>(
          wh[blk][1] + (size_t)g2 * 128 * 128, y2_h + (size_t)g2 * 128 * NPTS,
          P(5) + g2 * 128, P(6) + g2 * 128, P(7) + g2 * 128,
          nullptr, nullptr, gg_h + (size_t)g2 * 128 * NPTS,
          128, 512LL * NPTS, 512LL * NPTS, 1);
    }
    // fc2: 256x512, BN, +residual (x) -> x (in-place) f32 + f16
    gemm_bn_act<<<dim3(49, 2, BATCH), 256, 0, stream>>>(
        wh[blk][2], gg_h, P(9), P(10), P(11), x_f, x_f, x_h,
        512, 512LL * NPTS, 256LL * NPTS, 0);
    // ffn1: 1024x256, BN + GELU -> h f16
    gemm_bn_act<<<dim3(49, 8, BATCH), 256, 0, stream>>>(
        wh[blk][3], x_h, P(13), P(14), P(15), nullptr, nullptr, h_h,
        256, 256LL * NPTS, 1024LL * NPTS, 1);
    // ffn2: 256x1024, BN, +residual -> x (or d_out on last block)
    float*    yout = (blk == 1) ? (float*)d_out : x_f;
    _Float16* yh   = (blk == 1) ? nullptr : x_h;
    gemm_bn_act<<<dim3(49, 2, BATCH), 256, 0, stream>>>(
        wh[blk][4], h_h, P(17), P(18), P(19), x_f, yout, yh,
        1024, 1024LL * NPTS, 256LL * NPTS, 0);
  }
}
